// LSTMSignLanguageModel_23562190586298
// MI455X (gfx1250) — matrix operand order repaired and checked by reading, not compiled, not run on hardware
//
#include <hip/hip_runtime.h>
#include <hip/hip_bf16.h>

// ---------- types ----------
typedef __attribute__((ext_vector_type(16))) __bf16 v16bf;
typedef __attribute__((ext_vector_type(8)))  __bf16 v8bf;
typedef __attribute__((ext_vector_type(8)))  float  v8f;

static __device__ __forceinline__ __bf16 f2bf(float f) {
    union { float f; unsigned u; } v; v.f = f;
    unsigned r = v.u + 0x7FFFu + ((v.u >> 16) & 1u);   // round-to-nearest-even
    unsigned short s = (unsigned short)(r >> 16);
    return __builtin_bit_cast(__bf16, s);
}

static __device__ __forceinline__ v16bf load_a_frag(const __bf16* p) {
    v8bf r0 = *(const v8bf*)(p);        // 16B load
    v8bf r1 = *(const v8bf*)(p + 8);    // 16B load
    return __builtin_shufflevector(r0, r1,
        0, 1, 2, 3, 4, 5, 6, 7, 8, 9, 10, 11, 12, 13, 14, 15);
}

// ---------- WMMA inner product over one K-segment ----------
// A: bf16 row-major, MT*16 rows starting at Arow0 (lda elements between rows)
// Bt: bf16, [N][Kseg] (K contiguous). Wave computes MT x NT 16x16 tiles at
// rows m0 + mt*16 and columns nbase + t*16.
// Fragment layouts used here (contiguous k order, same for A and B):
//   A lane: M = lane%16, K = kk+16*half+[0..16)
//   B lane: N = nbase+t*16+lane%16, K = kk+16*half+[0..16)
// All loads for a K-step are hoisted ahead of the WMMA chain so the backend
// can clause them and overlap VMEM latency with XDL execution.
template <int MT, int NT>
static __device__ __forceinline__ void mma_seg(const __bf16* __restrict__ Arow0,
                                               int lda, int Kseg,
                                               const __bf16* __restrict__ Bt,
                                               int nbase, int lane,
                                               v8f (&acc)[MT][NT]) {
    const int half = lane >> 4;
    const int mrow = lane & 15;
    const __bf16* Arow = Arow0 + (size_t)mrow * lda;
    const __bf16* Brow = Bt + (size_t)mrow * Kseg + (half << 4);
    for (int kk = 0; kk < Kseg; kk += 32) {
        v16bf a[MT];
#pragma unroll
        for (int mt = 0; mt < MT; ++mt)
            a[mt] = load_a_frag(Arow + (size_t)(mt * 16) * lda + kk + half * 16);
        v16bf b[NT];
#pragma unroll
        for (int t = 0; t < NT; ++t)
            b[t] = *(const v16bf*)(Brow + (size_t)(nbase + t * 16) * Kseg + kk);
#pragma unroll
        for (int t = 0; t < NT; ++t)
#pragma unroll
            for (int mt = 0; mt < MT; ++mt)
                acc[mt][t] = __builtin_amdgcn_wmma_f32_16x16x32_bf16(
                    false, a[mt], false, b[t], (short)0, acc[mt][t], false, false);
    }
}

// ---------- fused LSTM step ----------
// z = A1 @ B1t^T + hprev @ B2t^T  (bias added in epilogue)
// grid: (4, 8) -> blockIdx.x = 128-col chunk of H, blockIdx.y = 32-row chunk of B
// block: 256 threads = 8 waves; wave w -> gate = w&3, nGroup = w>>2
// wave computes 2 M-tiles x 4 N-tiles (32 rows x 64 cols of one gate)
__global__ void lstm_step_kernel(const __bf16* __restrict__ A1, int lda1, int K1,
                                 const __bf16* __restrict__ B1t,   // [2048][K1]
                                 const __bf16* __restrict__ hprev, // [256][512] bf16
                                 const __bf16* __restrict__ B2t,   // [2048][512]
                                 const float* __restrict__ bias,   // [2048]
                                 float* __restrict__ c,            // [256][512] f32 in-place
                                 __bf16* __restrict__ hout) {      // [256][512] bf16
    __shared__ float zs[4][32][132];
    const int lane = threadIdx.x & 31;
    const int wave = threadIdx.x >> 5;
    const int gate = wave & 3;
    const int nGroup = wave >> 2;
    const int bx = blockIdx.x, by = blockIdx.y;
    const int m0 = by * 32;
    const int nbase = gate * 512 + bx * 128 + nGroup * 64;

    v8f acc[2][4] = {};
    mma_seg<2, 4>(A1 + (size_t)m0 * lda1, lda1, K1, B1t, nbase, lane, acc);
    mma_seg<2, 4>(hprev + (size_t)m0 * 512, 512, 512, B2t, nbase, lane, acc);

    const int half = lane >> 4, mrow = lane & 15;
#pragma unroll
    for (int mt = 0; mt < 2; ++mt)
#pragma unroll
        for (int t = 0; t < 4; ++t)
#pragma unroll
            for (int v = 0; v < 8; ++v)
                zs[gate][mt * 16 + v + 8 * half][nGroup * 64 + t * 16 + mrow] =
                    acc[mt][t][v];
    __syncthreads();

    const int tid = threadIdx.x;
#pragma unroll
    for (int e = 0; e < 16; ++e) {
        int o = tid + e * 256;
        int r = o >> 7, cc = o & 127;
        int brow = by * 32 + r;
        int j = bx * 128 + cc;
        float zi = zs[0][r][cc] + bias[j];
        float zf = zs[1][r][cc] + bias[512 + j];
        float zg = zs[2][r][cc] + bias[1024 + j];
        float zo = zs[3][r][cc] + bias[1536 + j];
        float ig = 1.f / (1.f + __expf(-zi));
        float fg = 1.f / (1.f + __expf(-zf));
        float gg = tanhf(zg);
        float og = 1.f / (1.f + __expf(-zo));
        size_t idx = (size_t)brow * 512 + j;
        float cn = fg * c[idx] + ig * gg;
        c[idx] = cn;
        hout[idx] = f2bf(og * tanhf(cn));
    }
}

// ---------- head GEMM: C = act(A @ Bt^T + bias) ----------
// grid: (N/128, M/16); block 256 threads, wave w handles one 16x16 tile.
// outBf16 != 0 -> write bf16 (for chained GEMM A input), else f32.
__global__ void gemm_head_kernel(const __bf16* __restrict__ A, int lda, int K,
                                 const __bf16* __restrict__ Bt,  // [Npad][K]
                                 const float* __restrict__ bias, int biasN,
                                 void* __restrict__ C, int ldc,
                                 int doRelu, int outBf16) {
    const int lane = threadIdx.x & 31;
    const int wave = threadIdx.x >> 5;
    const int m0 = blockIdx.y * 16;
    const int n0 = blockIdx.x * 128 + wave * 16;
    v8f acc[1][1] = {};
    mma_seg<1, 1>(A + (size_t)m0 * lda, lda, K, Bt, n0, lane, acc);
    const int half = lane >> 4, mrow = lane & 15;
    const int col = n0 + mrow;
    const float bb = (col < biasN) ? bias[col] : 0.f;
#pragma unroll
    for (int v = 0; v < 8; ++v) {
        float val = acc[0][0][v] + bb;
        if (doRelu) val = fmaxf(val, 0.f);
        size_t idx = (size_t)(m0 + v + 8 * half) * ldc + col;
        if (outBf16) ((__bf16*)C)[idx] = f2bf(val);
        else         ((float*)C)[idx] = val;
    }
}

// ---------- softmax over 500 classes, one block per row ----------
__global__ void softmax500_kernel(const float* __restrict__ logits, int ldl,
                                  float* __restrict__ out) {
    __shared__ float red[256];
    const int row = blockIdx.x, tid = threadIdx.x;
    const float* lr = logits + (size_t)row * ldl;
    float m = -1e30f;
    for (int j = tid; j < 500; j += 256) m = fmaxf(m, lr[j]);
    red[tid] = m; __syncthreads();
    for (int s = 128; s > 0; s >>= 1) { if (tid < s) red[tid] = fmaxf(red[tid], red[tid + s]); __syncthreads(); }
    m = red[0]; __syncthreads();
    float sum = 0.f;
    for (int j = tid; j < 500; j += 256) sum += __expf(lr[j] - m);
    red[tid] = sum; __syncthreads();
    for (int s = 128; s > 0; s >>= 1) { if (tid < s) red[tid] += red[tid + s]; __syncthreads(); }
    const float inv = 1.f / red[0];
    for (int j = tid; j < 500; j += 256) out[(size_t)row * 500 + j] = __expf(lr[j] - m) * inv;
}

// ---------- weight transpose + f32->bf16 convert: dst[n][k] = src[k][n] ----------
__global__ void convT_kernel(const float* __restrict__ src, __bf16* __restrict__ dst,
                             int K, int N, int Npad) {
    size_t idx = (size_t)blockIdx.x * blockDim.x + threadIdx.x;
    size_t tot = (size_t)Npad * K;
    if (idx >= tot) return;
    int n = (int)(idx / K), k = (int)(idx % K);
    dst[idx] = (n < N) ? f2bf(src[(size_t)k * N + n]) : f2bf(0.f);
}

// ---------- straight f32 -> bf16 copy (same layout) ----------
__global__ void conv_kernel(const float* __restrict__ src, __bf16* __restrict__ dst,
                            size_t n) {
    size_t i = (size_t)blockIdx.x * blockDim.x + threadIdx.x;
    if (i < n) dst[i] = f2bf(src[i]);
}

// ---------- zero bytes (16B granules) ----------
__global__ void zerob_kernel(uint4* __restrict__ p, size_t n16) {
    size_t i = (size_t)blockIdx.x * blockDim.x + threadIdx.x;
    if (i < n16) p[i] = uint4{0u, 0u, 0u, 0u};
}

// ---------- launch ----------
extern "C" void kernel_launch(void* const* d_in, const int* in_sizes, int n_in,
                              void* d_out, int out_size, void* d_ws, size_t ws_size,
                              hipStream_t stream) {
    (void)in_sizes; (void)n_in; (void)out_size; (void)ws_size;
    const float* x   = (const float*)d_in[0];   // (256,256,256)
    const float* Wx1 = (const float*)d_in[1];   // (256,2048)
    const float* Wh1 = (const float*)d_in[2];   // (512,2048)
    const float* b1  = (const float*)d_in[3];
    const float* Wx2 = (const float*)d_in[4];   // (512,2048)
    const float* Wh2 = (const float*)d_in[5];
    const float* b2  = (const float*)d_in[6];
    const float* Wd  = (const float*)d_in[7];   // (512,512)
    const float* bd  = (const float*)d_in[8];
    const float* Wo  = (const float*)d_in[9];   // (512,500)
    const float* bo  = (const float*)d_in[10];

    size_t off = 0;
    char* base = (char*)d_ws;
    auto alloc = [&](size_t bytes) -> void* {
        void* p = base + off;
        off += (bytes + 255) & ~(size_t)255;
        return p;
    };

    __bf16* Wx1t = (__bf16*)alloc((size_t)2048 * 256 * 2);
    __bf16* Wh1t = (__bf16*)alloc((size_t)2048 * 512 * 2);
    __bf16* Wx2t = (__bf16*)alloc((size_t)2048 * 512 * 2);
    __bf16* Wh2t = (__bf16*)alloc((size_t)2048 * 512 * 2);
    __bf16* Wdt  = (__bf16*)alloc((size_t)512  * 512 * 2);
    __bf16* Wot  = (__bf16*)alloc((size_t)512  * 512 * 2);

    const size_t XN = (size_t)256 * 256 * 256;    // x element count
    __bf16* xbf = (__bf16*)alloc(XN * 2);         // bf16 copy of x

    const size_t S = (size_t)256 * 512;           // one state buffer (elements)
    // contiguous zero-init region: h1a(bf16), h2a(bf16), c1(f32), c2(f32)
    __bf16* h1a  = (__bf16*)alloc(S * 2);
    __bf16* h2a  = (__bf16*)alloc(S * 2);
    float*  c1   = (float*)alloc(S * 4);
    float*  c2   = (float*)alloc(S * 4);
    __bf16* h1b  = (__bf16*)alloc(S * 2);
    __bf16* h2b  = (__bf16*)alloc(S * 2);
    __bf16* dbuf = (__bf16*)alloc(S * 2);         // relu dense out (256x512) bf16
    float*  logit = (float*)alloc(S * 4);         // padded logits (256x512) f32

    const int thr = 256;

    // 1) weight conversion / transpose + x conversion
    convT_kernel<<<(2048 * 256 + thr - 1) / thr, thr, 0, stream>>>(Wx1, Wx1t, 256, 2048, 2048);
    convT_kernel<<<(2048 * 512 + thr - 1) / thr, thr, 0, stream>>>(Wh1, Wh1t, 512, 2048, 2048);
    convT_kernel<<<(2048 * 512 + thr - 1) / thr, thr, 0, stream>>>(Wx2, Wx2t, 512, 2048, 2048);
    convT_kernel<<<(2048 * 512 + thr - 1) / thr, thr, 0, stream>>>(Wh2, Wh2t, 512, 2048, 2048);
    convT_kernel<<<(512 * 512 + thr - 1) / thr, thr, 0, stream>>>(Wd, Wdt, 512, 512, 512);
    convT_kernel<<<(512 * 512 + thr - 1) / thr, thr, 0, stream>>>(Wo, Wot, 512, 500, 512);
    conv_kernel<<<(unsigned)((XN + thr - 1) / thr), thr, 0, stream>>>(x, xbf, XN);

    // 2) zero initial states: h1a, h2a (bf16) + c1, c2 (f32) contiguous
    {
        size_t bytes = S * 2 + S * 2 + S * 4 + S * 4;   // 1.5 MB
        size_t n16 = bytes / 16;
        zerob_kernel<<<(unsigned)((n16 + thr - 1) / thr), thr, 0, stream>>>((uint4*)h1a, n16);
    }

    // 3) sequential scan: fused per-step kernels, ping-pong h buffers
    __bf16* h1buf[2] = {h1a, h1b};
    __bf16* h2buf[2] = {h2a, h2b};
    for (int t = 0; t < 256; ++t) {
        const __bf16* xA = xbf + (size_t)t * 256;  // row stride T*256 = 65536
        lstm_step_kernel<<<dim3(4, 8), 256, 0, stream>>>(
            xA, 65536, 256, Wx1t, h1buf[t & 1], Wh1t, b1, c1, h1buf[(t + 1) & 1]);
        lstm_step_kernel<<<dim3(4, 8), 256, 0, stream>>>(
            h1buf[(t + 1) & 1], 512, 512, Wx2t, h2buf[t & 1], Wh2t, b2, c2,
            h2buf[(t + 1) & 1]);
    }
    __bf16* h2_final = h2buf[0];  // after t=255, write index was (255+1)&1 == 0

    // 4) head: relu(h2 @ Wd + bd) -> bf16, then logits = d @ Wo + bo -> f32 (padded N=512)
    gemm_head_kernel<<<dim3(4, 16), 256, 0, stream>>>(h2_final, 512, 512, Wdt, bd, 512, dbuf, 512, 1, 1);
    gemm_head_kernel<<<dim3(4, 16), 256, 0, stream>>>(dbuf, 512, 512, Wot, bo, 500, logit, 512, 0, 0);

    // 5) softmax over the 500 real classes
    softmax500_kernel<<<256, 256, 0, stream>>>(logit, 512, (float*)d_out);
}
